// AF2dMADEBlock_50663434224127
// MI455X (gfx1250) — matrix operand order repaired and checked by reading, not compiled, not run on hardware
//
#include <hip/hip_runtime.h>
#include <hip/hip_bf16.h>
#include <math.h>

typedef _Float16 h16;
typedef h16  v16h __attribute__((ext_vector_type(16)));
typedef h16  v8h  __attribute__((ext_vector_type(8)));
typedef float v8f __attribute__((ext_vector_type(8)));

// Problem constants: B=16, C=3, H=W=8 (64 pixels), HID=64, 3x3 kernels,
// nets: mu, lv; layers: w0 (mask A), w1, w2 (mask B hidden), w3 (mask B out).
// Masked taps (di,dj):
//   mask A: (-1,-1),(-1,0),(-1,1),(0,-1)           -> K0 = 4*3  = 12
//   mask B: (-1,-1),(-1,0),(-1,1),(0,-1),(0,0)     -> KH = 5*64 = 320

// ---------------- LDS layout (dynamic shared memory) ----------------
constexpr int WH_OFF  = 0;
constexpr int WH_SZ   = 2*2*64*320*2;     // f16 wh[net][lay][cout][k]
constexpr int IMA_OFF = WH_OFF + WH_SZ;   // f16 im2col A [pixel][320]
constexpr int IM_SZ   = 64*320*2;
constexpr int IMB_OFF = IMA_OFF + IM_SZ;  // f16 im2col B [pixel][320]
constexpr int W0_OFF  = IMB_OFF + IM_SZ;  // f16 w0[net][64][12]
constexpr int W0_SZ   = 2*64*12*2;
constexpr int W3_OFF  = W0_OFF + W0_SZ;   // f16 w3[net][3][320]
constexpr int W3_SZ   = 2*3*320*2;
constexpr int B0_OFF  = W3_OFF + W3_SZ;   // f32 [net][64]
constexpr int B1_OFF  = B0_OFF + 512;
constexpr int B2_OFF  = B1_OFF + 512;
constexpr int B3_OFF  = B2_OFF + 512;     // f32 [net][4]
constexpr int Y_OFF   = B3_OFF + 32;      // f32 y[3][64]
constexpr int X_OFF   = Y_OFF + 768;      // f32 x[3][64]
constexpr int O3_OFF  = X_OFF + 768;      // f32 out3[net][4]
constexpr int SMEM_BYTES = O3_OFF + 32;   // ~250 KB < 320 KB WGP LDS

__device__ __forceinline__ float eluf(float v) {
  return v > 0.f ? v : (__expf(v) - 1.f);
}

// One hidden masked-conv layer as implicit GEMM via v_wmma_f32_16x16x32_f16.
// W: [64 cout][320 k] f16 (LDS), Xim: [64 px][320 k] f16 (LDS im2col input),
// Yim: im2col destination for next layer (scatter with mask-B taps).
__device__ __forceinline__ void gemm_hidden(const h16* __restrict__ W,
                                            const h16* __restrict__ Xim,
                                            h16*       __restrict__ Yim,
                                            const float* __restrict__ bias,
                                            int wave, int lane)
{
  const int l15 = lane & 15;
  const int hiA = (lane < 16) ? 0 : 16;   // A layout (matches B): K 0-15 vs 16-31
  const int hiB = (lane < 16) ? 0 : 16;   // B layout (contiguous, matches A): K 0-15 vs 16-31
  #pragma unroll
  for (int r = 0; r < 2; ++r) {
    const int t  = wave * 2 + r;          // 8 waves x 2 = 16 tiles (4x4 of 16x16)
    const int mt = t >> 2, nt = t & 3;
    const int m  = mt * 16 + l15;         // cout row
    const int n  = nt * 16 + l15;         // pixel column
    const h16* wr = W   + m * 320;
    const h16* xr = Xim + n * 320;
    v8f acc = {};
    #pragma unroll
    for (int kc = 0; kc < 10; ++kc) {     // K = 320 = 10 x 32
      const int kb = kc * 32;
      v8h a0 = *(const v8h*)(wr + kb + hiA);
      v8h a1 = *(const v8h*)(wr + kb + hiA + 8);
      v16h A = __builtin_shufflevector(a0, a1, 0,1,2,3,4,5,6,7,8,9,10,11,12,13,14,15);
      v16h Bv = *(const v16h*)(xr + kb + hiB);
      acc = __builtin_amdgcn_wmma_f32_16x16x32_f16(false, A, false, Bv,
                                                   (short)0, acc, false, false);
    }
    // C layout: VGPR g, lanes 0-15 -> M=g, lanes 16-31 -> M=8+g; N = lane&15.
    const int mbase = mt * 16 + ((lane < 16) ? 0 : 8);
    const int q  = n;
    const int qi = q >> 3, qj = q & 7;
    constexpr int BDI[5] = {-1,-1,-1, 0, 0};
    constexpr int BDJ[5] = {-1, 0, 1,-1, 0};
    #pragma unroll
    for (int g = 0; g < 8; ++g) {
      const int ch = mbase + g;
      float hv = eluf(acc[g] + bias[ch]);
      h16 hh = (h16)hv;
      #pragma unroll
      for (int tp = 0; tp < 5; ++tp) {    // scatter h(q) into im2col rows using it
        int ri = qi - BDI[tp], rj = qj - BDJ[tp];
        if ((unsigned)ri < 8u && (unsigned)rj < 8u)
          Yim[(ri * 8 + rj) * 320 + tp * 64 + ch] = hh;
      }
    }
  }
}

__global__ __launch_bounds__(256, 1)
void af2d_made_scan_kernel(const float* __restrict__ x,
    const float* mu_w0, const float* mu_b0, const float* mu_w1, const float* mu_b1,
    const float* mu_w2, const float* mu_b2, const float* mu_w3, const float* mu_b3,
    const float* lv_w0, const float* lv_b0, const float* lv_w1, const float* lv_b1,
    const float* lv_w2, const float* lv_b2, const float* lv_w3, const float* lv_b3,
    float* __restrict__ out)
{
  extern __shared__ unsigned char smem[];
  h16*   wh  = (h16*)(smem + WH_OFF);
  h16*   imA = (h16*)(smem + IMA_OFF);
  h16*   imB = (h16*)(smem + IMB_OFF);
  h16*   w0l = (h16*)(smem + W0_OFF);
  h16*   w3l = (h16*)(smem + W3_OFF);
  float* b0l = (float*)(smem + B0_OFF);
  float* b1l = (float*)(smem + B1_OFF);
  float* b2l = (float*)(smem + B2_OFF);
  float* b3l = (float*)(smem + B3_OFF);
  float* ysh = (float*)(smem + Y_OFF);
  float* xsh = (float*)(smem + X_OFF);
  float* o3  = (float*)(smem + O3_OFF);

  const int b    = blockIdx.x;     // one workgroup per batch element
  const int tid  = threadIdx.x;
  const int lane = tid & 31;
  const int wave = tid >> 5;

  constexpr int BDI[5] = {-1,-1,-1, 0, 0};
  constexpr int BDJ[5] = {-1, 0, 1,-1, 0};
  constexpr int ADI[4] = {-1,-1,-1, 0};
  constexpr int ADJ[4] = {-1, 0, 1,-1};

  // ---------------- one-time weight pack (f32 global -> f16 LDS) ----------------
  for (int idx = tid; idx < 2*2*64*320; idx += 256) {   // hidden layers, masked taps only
    int k   = idx % 320;
    int co  = (idx / 320) % 64;
    int lay = (idx / (320*64)) % 2;
    int net = idx / (320*64*2);
    int tp = k / 64, ci = k % 64;
    const float* src = net ? (lay ? lv_w2 : lv_w1) : (lay ? mu_w2 : mu_w1);
    wh[idx] = (h16)src[((co*64 + ci)*3 + (BDI[tp]+1))*3 + (BDJ[tp]+1)];
  }
  for (int idx = tid; idx < 2*64*12; idx += 256) {      // input layer, mask A
    int ci  = idx % 3;
    int tp  = (idx / 3) % 4;
    int co  = (idx / 12) % 64;
    int net = idx / 768;
    const float* src = net ? lv_w0 : mu_w0;
    w0l[idx] = (h16)src[((co*3 + ci)*3 + (ADI[tp]+1))*3 + (ADJ[tp]+1)];
  }
  for (int idx = tid; idx < 2*3*320; idx += 256) {      // output layer, mask B
    int k   = idx % 320;
    int co  = (idx / 320) % 3;
    int net = idx / 960;
    int tp = k / 64, ci = k % 64;
    const float* src = net ? lv_w3 : mu_w3;
    w3l[idx] = (h16)src[((co*64 + ci)*3 + (BDI[tp]+1))*3 + (BDJ[tp]+1)];
  }
  for (int idx = tid; idx < 128; idx += 256) {
    int net = idx / 64, c = idx % 64;
    b0l[idx] = (net ? lv_b0 : mu_b0)[c];
    b1l[idx] = (net ? lv_b1 : mu_b1)[c];
    b2l[idx] = (net ? lv_b2 : mu_b2)[c];
  }
  if (tid < 8) {
    int net = tid / 4, c = tid % 4;
    b3l[tid] = (c < 3) ? (net ? lv_b3 : mu_b3)[c] : 0.f;
  }
  for (int idx = tid; idx < 2*64*320; idx += 256)       // zero both im2col buffers
    imA[idx] = (h16)0.f;                                // (imB is contiguous after imA)
  for (int idx = tid; idx < 192; idx += 256) {          // load x, zero y
    xsh[idx] = x[b * 192 + idx];
    ysh[idx] = 0.f;
  }
  __syncthreads();

  float lsum = 0.f;  // accumulated by tid 0 (== final-map logstd sum by causality)

  // ---------------- raster scan over 64 pixels ----------------
  for (int p = 0; p < 64; ++p) {
    for (int net = 0; net < 2; ++net) {
      const h16* whn = wh + (net * 2) * 64 * 320;

      // layer 0 (3 -> 64, mask A) on VALU; write elu result into im2col A
      for (int idx = tid; idx < 64 * 64; idx += 256) {
        int ch = idx >> 6, q = idx & 63;
        int qi = q >> 3, qj = q & 7;
        float acc = b0l[net * 64 + ch];
        #pragma unroll
        for (int tp = 0; tp < 4; ++tp) {
          int ni = qi + ADI[tp], nj = qj + ADJ[tp];
          if ((unsigned)ni < 8u && (unsigned)nj < 8u) {
            int nq = ni * 8 + nj;
            #pragma unroll
            for (int ci = 0; ci < 3; ++ci)
              acc += (float)w0l[(net * 64 + ch) * 12 + tp * 3 + ci] * ysh[ci * 64 + nq];
          }
        }
        h16 hh = (h16)eluf(acc);
        #pragma unroll
        for (int tp = 0; tp < 5; ++tp) {
          int ri = qi - BDI[tp], rj = qj - BDJ[tp];
          if ((unsigned)ri < 8u && (unsigned)rj < 8u)
            imA[(ri * 8 + rj) * 320 + tp * 64 + ch] = hh;
        }
      }
      __syncthreads();

      // hidden layer 1: WMMA GEMM, imA -> imB
      gemm_hidden(whn,            imA, imB, b1l + net * 64, wave, lane);
      __syncthreads();
      // hidden layer 2: WMMA GEMM, imB -> imA
      gemm_hidden(whn + 64 * 320, imB, imA, b2l + net * 64, wave, lane);
      __syncthreads();

      // output layer (64 -> 3, mask B): only pixel p needed; one wave per channel
      if (wave < 3) {
        const h16* col = imA + p * 320;
        const h16* wr  = w3l + (net * 3 + wave) * 320;
        float acc = 0.f;
        for (int k = lane; k < 320; k += 32)
          acc += (float)wr[k] * (float)col[k];
        #pragma unroll
        for (int off = 16; off > 0; off >>= 1)
          acc += __shfl_xor(acc, off, 32);
        if (lane == 0) o3[net * 4 + wave] = acc + b3l[net * 4 + wave];
      }
      __syncthreads();
    }

    // pixel update: new_y = (x - mu) / (exp(0.5*lv) + eps)
    if (tid == 0) {
      #pragma unroll
      for (int c = 0; c < 3; ++c) {
        float mu     = o3[c];
        float logstd = 0.5f * o3[4 + c];
        lsum += logstd;
        ysh[c * 64 + p] = (xsh[c * 64 + p] - mu) / (__expf(logstd) + 1e-12f);
      }
    }
    __syncthreads();
  }

  // ---------------- write outputs: y [16,3,8,8] then logstd sums [16] ----------------
  for (int idx = tid; idx < 192; idx += 256)
    out[b * 192 + idx] = ysh[idx];
  if (tid == 0)
    out[16 * 192 + b] = lsum;
}

extern "C" void kernel_launch(void* const* d_in, const int* in_sizes, int n_in,
                              void* d_out, int out_size, void* d_ws, size_t ws_size,
                              hipStream_t stream) {
  (void)in_sizes; (void)n_in; (void)d_ws; (void)ws_size; (void)out_size;
  const float* x     = (const float*)d_in[0];
  const float* mu_w0 = (const float*)d_in[1];
  const float* mu_b0 = (const float*)d_in[2];
  const float* mu_w1 = (const float*)d_in[3];
  const float* mu_b1 = (const float*)d_in[4];
  const float* mu_w2 = (const float*)d_in[5];
  const float* mu_b2 = (const float*)d_in[6];
  const float* mu_w3 = (const float*)d_in[7];
  const float* mu_b3 = (const float*)d_in[8];
  const float* lv_w0 = (const float*)d_in[9];
  const float* lv_b0 = (const float*)d_in[10];
  const float* lv_w1 = (const float*)d_in[11];
  const float* lv_b1 = (const float*)d_in[12];
  const float* lv_w2 = (const float*)d_in[13];
  const float* lv_b2 = (const float*)d_in[14];
  const float* lv_w3 = (const float*)d_in[15];
  const float* lv_b3 = (const float*)d_in[16];
  float* out = (float*)d_out;

  af2d_made_scan_kernel<<<16, 256, SMEM_BYTES, stream>>>(
      x, mu_w0, mu_b0, mu_w1, mu_b1, mu_w2, mu_b2, mu_w3, mu_b3,
      lv_w0, lv_b0, lv_w1, lv_b1, lv_w2, lv_b2, lv_w3, lv_b3, out);
}
